// CBAMUNet_MICCAI_9070970929766
// MI455X (gfx1250) — compile-verified, matrix operand order checked by reading, not run on hardware
//
#include <hip/hip_runtime.h>
#include <math.h>

typedef __attribute__((ext_vector_type(16))) _Float16 v16h;
typedef __attribute__((ext_vector_type(8)))  _Float16 v8h;
typedef __attribute__((ext_vector_type(8)))  float    v8f;

#define BATCH 8
#define HH 128
#define WW 128

// ---------------------------------------------------------------------------
// WMMA fragment helpers (CDNA5 16x16x32 f16, wave32).  ISA 7.12.2:
//  A (16x32 f16): lane L holds row m=L%16; VGPR v packs K pair
//    k = (v<4 ? 2v : 8+2v) + 8*(L/16).
//  B (32x16 f16): lane L holds col n=L%16, same K mapping.
//  C/D (16x16 f32): lane L holds col n=L%16, VGPR v holds row m=v+8*(L/16).
// A lane therefore needs k in [8h,8h+8) and [16+8h,16+8h+8): when the 32
// K-values are contiguous in memory this is exactly two 16-byte loads.
// ---------------------------------------------------------------------------
__device__ __forceinline__ int amap_k(int v, int h) { return (v < 4 ? 2 * v : 8 + 2 * v) + 8 * h; }

template <typename Fn>
__device__ __forceinline__ v16h build_frag(Fn f) {
  const int L = threadIdx.x & 31;
  const int h = L >> 4;
  v16h r;
#pragma unroll
  for (int v = 0; v < 8; ++v) {
    int k = amap_k(v, h);
    r[2 * v]     = f(k);
    r[2 * v + 1] = f(k + 1);
  }
  return r;
}

// Fragment from 32 contiguous (16B-aligned) K-values: two b128 loads.
__device__ __forceinline__ v16h frag_contig(const _Float16* p, int h) {
  v8h lo = *(const v8h*)(p + 8 * h);
  v8h hi = *(const v8h*)(p + 16 + 8 * h);
  v16h r;
#pragma unroll
  for (int i = 0; i < 8; ++i) { r[i] = lo[i]; r[8 + i] = hi[i]; }
  return r;
}

__device__ __forceinline__ v8f wmma_f16(v16h a, v16h b, v8f c) {
  return __builtin_amdgcn_wmma_f32_16x16x32_f16(false, a, false, b, (short)0, c, false, false);
}

// Keys cubic, a = -0.5 (matches jax.image.resize 'cubic')
__device__ __forceinline__ float cubicw(float x) {
  x = fabsf(x);
  if (x < 1.f) return ((1.5f * x - 2.5f) * x) * x + 1.f;
  if (x < 2.f) return ((-0.5f * x + 2.5f) * x - 4.f) * x + 2.f;
  return 0.f;
}

// ---------------------------------------------------------------------------
__global__ void k_init_out(const float* __restrict__ x, float* __restrict__ out) {
  int i = blockIdx.x * 256 + threadIdx.x;
  if (i < BATCH * HH * WW) out[i] = x[i];
}

// ---------------------------------------------------------------------------
// Gate conv1: 3x3, 1->64, ReLU.  Channel-last f16 [B,H,W,64] so the implicit
// GEMM K index (tap*64+ch) is contiguous per 32-wide K window.
// ---------------------------------------------------------------------------
__global__ void k_gate_c1(const float* __restrict__ x, const float* __restrict__ w1,
                          const float* __restrict__ b1, _Float16* __restrict__ h1) {
  int idx = blockIdx.x * 256 + threadIdx.x;
  if (idx >= BATCH * HH * WW) return;
  int px = idx & 127, py = (idx >> 7) & 127, b = idx >> 14;
  float in[9];
#pragma unroll
  for (int t = 0; t < 9; ++t) {
    int dy = t / 3 - 1, dx = t % 3 - 1;
    int yy = py + dy, xx = px + dx;
    in[t] = ((unsigned)yy < 128u && (unsigned)xx < 128u) ? x[(b << 14) + (yy << 7) + xx] : 0.f;
  }
  _Float16* dst = h1 + ((long)idx << 6);
  for (int c = 0; c < 64; ++c) {
    float a = b1[c];
#pragma unroll
    for (int t = 0; t < 9; ++t) a += w1[c * 9 + t] * in[t];
    dst[c] = (_Float16)fmaxf(a, 0.f);
  }
}

// w2 (64,64,3,3) -> transposed w2rT[n=outch][k=tap*64+ch] f16 (64x576)
__global__ void k_prep_w2r(const float* __restrict__ w2, _Float16* __restrict__ w2rT) {
  int idx = blockIdx.x * 256 + threadIdx.x;
  if (idx >= 576 * 64) return;
  int n = idx & 63, kk = idx >> 6;
  int tap = kk >> 6, ch = kk & 63;
  int dy = tap / 3, dx = tap % 3;
  w2rT[n * 576 + kk] = (_Float16)w2[((n * 64 + ch) * 3 + dy) * 3 + dx];
}

// c1w (7,64,1,5,5) -> transposed c1wrT[e][outch][k(pad 25->32)] f16
__global__ void k_prep_c1wr(const float* __restrict__ c1w, _Float16* __restrict__ c1wrT) {
  int idx = blockIdx.x * 256 + threadIdx.x;
  if (idx >= 7 * 32 * 64) return;
  int kk = idx & 31, ch = (idx >> 5) & 63, e = idx >> 11;
  float v = (kk < 25) ? c1w[e * 1600 + ch * 25 + kk] : 0.f;
  c1wrT[(e * 64 + ch) * 32 + kk] = (_Float16)v;
}

// ---------------------------------------------------------------------------
// Gate conv2 (3x3 64->64 implicit GEMM via WMMA, K=576) fused with ReLU,
// conv1x1 64->7, softmax over 7, top-4 mask.  One wave per 16-pixel tile.
// ---------------------------------------------------------------------------
__global__ void k_gate_c2(const _Float16* __restrict__ h1, const _Float16* __restrict__ w2rT,
                          const float* __restrict__ b2, const float* __restrict__ w3,
                          const float* __restrict__ b3, float* __restrict__ gw) {
  int tile = blockIdx.x;                 // B*H*(W/16) tiles
  int x0 = (tile % (WW / 16)) * 16;
  int y  = (tile / (WW / 16)) % HH;
  int b  = tile / ((WW / 16) * HH);
  const int L = threadIdx.x & 31, h = L >> 4, m = L & 15, n = L & 15;

  v8f acc[4] = {};
  for (int k0 = 0; k0 < 576; k0 += 32) {
    v16h A = {};
    {
      int tap = k0 >> 6, ch0 = k0 & 63;
      int dy = tap / 3 - 1, dx = tap % 3 - 1;
      int yy = y + dy, xx = x0 + m + dx;
      if ((unsigned)yy < (unsigned)HH && (unsigned)xx < (unsigned)WW)
        A = frag_contig(h1 + (((long)(b * HH + yy) * WW + xx) << 6) + ch0, h);
    }
#pragma unroll
    for (int t = 0; t < 4; ++t) {
      v16h B = frag_contig(w2rT + (t * 16 + n) * 576 + k0, h);
      acc[t] = wmma_f16(A, B, acc[t]);
    }
  }

  __shared__ float sh[16 * 64];
#pragma unroll
  for (int t = 0; t < 4; ++t)
#pragma unroll
    for (int v = 0; v < 8; ++v) {
      int mm = v + 8 * h, nn = t * 16 + n;
      float val = acc[t][v] + b2[nn];
      sh[mm * 64 + nn] = val > 0.f ? val : 0.f;
    }
  __syncthreads();

  if (L < 16) {  // one lane per pixel: 1x1 conv -> softmax -> top-4 mask
    float logit[7];
#pragma unroll
    for (int e = 0; e < 7; ++e) logit[e] = b3[e];
    for (int c = 0; c < 64; ++c) {
      float hv = sh[L * 64 + c];
#pragma unroll
      for (int e = 0; e < 7; ++e) logit[e] += w3[e * 64 + c] * hv;
    }
    float mx = logit[0];
#pragma unroll
    for (int e = 1; e < 7; ++e) mx = fmaxf(mx, logit[e]);
    float p[7], se = 0.f;
#pragma unroll
    for (int e = 0; e < 7; ++e) { p[e] = __expf(logit[e] - mx); se += p[e]; }
    float inv = 1.f / se;
#pragma unroll
    for (int e = 0; e < 7; ++e) p[e] *= inv;
    float tmp[7], thr = 0.f;
#pragma unroll
    for (int e = 0; e < 7; ++e) tmp[e] = p[e];
    for (int r = 0; r < 4; ++r) {
      int bi = 0; float bv = -1.f;
#pragma unroll
      for (int e = 0; e < 7; ++e) if (tmp[e] > bv) { bv = tmp[e]; bi = e; }
      thr = bv; tmp[bi] = -2.f;
    }
#pragma unroll
    for (int e = 0; e < 7; ++e)
      gw[(((long)(b * 7 + e) * HH + y) << 7) + x0 + L] = (p[e] >= thr) ? p[e] : 0.f;
  }
}

// ---------------------------------------------------------------------------
// Generic separable cubic resize (one axis), antialiased for downscale.
// ---------------------------------------------------------------------------
__global__ void k_resize1d(const float* __restrict__ src, float* __restrict__ dst,
                           int n_b, int n_keep, int in_len, int out_len,
                           long sb, long sk, long se, long db, long dk, long de) {
  long idx = (long)blockIdx.x * 256 + threadIdx.x;
  long tot = (long)n_b * n_keep * out_len;
  if (idx >= tot) return;
  int o = (int)(idx % out_len);
  long r = idx / out_len;
  int kp = (int)(r % n_keep);
  int b  = (int)(r / n_keep);
  float inv = (float)in_len / (float)out_len;
  float ks  = inv > 1.f ? inv : 1.f;
  float c   = (o + 0.5f) * inv - 0.5f;
  int lo = (int)ceilf(c - 2.f * ks);
  int hi = (int)floorf(c + 2.f * ks);
  const float* s = src + (long)b * sb + (long)kp * sk;
  float sum = 0.f, wsum = 0.f;
  for (int i = lo; i <= hi; ++i) {
    if ((unsigned)i >= (unsigned)in_len) continue;
    float w = cubicw((i - c) / ks);
    sum += w * s[(long)i * se];
    wsum += w;
  }
  dst[(long)b * db + (long)kp * dk + (long)o * de] = sum / wsum;
}

// ---------------------------------------------------------------------------
// q/k = affine(xi) f16, and v stored TRANSPOSED (vT[j][p] = v[p][j]) so the
// P.V B-operand is K-contiguous.  Buffers zero-padded to [B, sp, sp].
// ---------------------------------------------------------------------------
__global__ void k_qkv(const float* __restrict__ xi,
                      _Float16* __restrict__ q, _Float16* __restrict__ k, _Float16* __restrict__ vT,
                      const float* qw, const float* qb, const float* kw, const float* kb,
                      const float* vw, const float* vb, int e, int s, int sp) {
  long idx = (long)blockIdx.x * 256 + threadIdx.x;
  long tot = (long)BATCH * sp * sp;
  if (idx >= tot) return;
  int jj = (int)(idx % sp);
  int ii = (int)((idx / sp) % sp);
  int b  = (int)(idx / ((long)sp * sp));
  _Float16 qv = (_Float16)0, kv = (_Float16)0, vv = (_Float16)0;
  if (ii < s && jj < s) {
    float xv = xi[((long)b * s + ii) * s + jj];
    qv = (_Float16)(xv * qw[e] + qb[e]);
    kv = (_Float16)(xv * kw[e] + kb[e]);
    vv = (_Float16)(xv * vw[e] + vb[e]);
  }
  q[idx] = qv; k[idx] = kv;
  vT[((long)b * sp + jj) * sp + ii] = vv;
}

// ---------------------------------------------------------------------------
// Single-channel attention + residual.  One wave owns a 16-row block.
// Q block staged to LDS via GLOBAL_LOAD_ASYNC_TO_LDS_B128 (ASYNCcnt path,
// no VGPR round-trip), scores via WMMA into LDS, row softmax, P f16 overlays
// the Q buffer, then P.V via WMMA with the transposed V.  The K loops use two
// independent accumulators so back-to-back v_wmma have no D-register RAW
// (ISA 7.12.1 hazard) and next-chunk b128 loads overlap the current WMMA.
// ---------------------------------------------------------------------------
__global__ void k_attn(const _Float16* __restrict__ qg, const _Float16* __restrict__ kg,
                       const _Float16* __restrict__ vTg, const float* __restrict__ xi,
                       float* __restrict__ hout, int s, int sp) {
  __shared__ float    sS[16 * 512];
  __shared__ _Float16 sQP[16 * 512];   // Q during phase 1, P during phases 2-3
  int b = blockIdx.x, rt = blockIdx.y;
  int i0 = rt * 16;
  const int L = threadIdx.x & 31, h = L >> 4, m = L & 15, n = L & 15;
  const long base = (long)b * sp * sp;
  const _Float16* qb = qg + base;
  const _Float16* kb = kg + base;
  const _Float16* vb = vTg + base;
  const int njt = sp / 16;

  // Stage the 16 x sp Q block into LDS with async b128 copies (per-lane
  // LDS offset = low 32 bits of the generic pointer).
  {
    int cpr = sp / 8;                       // 8-half (16B) chunks per row
    for (int cidx = L; cidx < 16 * cpr; cidx += 32) {
      int row = cidx / cpr, col = (cidx % cpr) * 8;
      unsigned lds = (unsigned)(unsigned long long)(void*)&sQP[row * sp + col];
      const _Float16* g = qb + (long)(i0 + row) * sp + col;
      asm volatile("global_load_async_to_lds_b128 %0, %1, off"
                   :: "v"(lds), "v"(g) : "memory");
    }
    asm volatile("s_wait_asynccnt 0x0" ::: "memory");
  }
  __syncthreads();

  // Phase 1: S = Q K^T   (sp is a multiple of 32; unroll 2 when possible)
  for (int jt = 0; jt < njt; ++jt) {
    v8f acc0 = {}, acc1 = {};
    const _Float16* krow = kb + (long)(jt * 16 + n) * sp;
    for (int c0 = 0; c0 < sp; c0 += 64) {
      __builtin_prefetch(krow + c0 + 64, 0, 3);
      v16h A0 = frag_contig(&sQP[m * sp + c0], h);
      v16h B0 = frag_contig(krow + c0, h);
      acc0 = wmma_f16(A0, B0, acc0);
      if (c0 + 32 < sp) {
        v16h A1 = frag_contig(&sQP[m * sp + c0 + 32], h);
        v16h B1 = frag_contig(krow + c0 + 32, h);
        acc1 = wmma_f16(A1, B1, acc1);
      }
    }
#pragma unroll
    for (int v = 0; v < 8; ++v) sS[(v + 8 * h) * sp + jt * 16 + n] = acc0[v] + acc1[v];
  }
  __syncthreads();

  // Phase 2: row softmax (lanes 0..15, one row each), P f16 over the Q buffer
  if (h == 0) {
    int row = m;
    float mxv = -1e30f;
    for (int j = 0; j < s; ++j) mxv = fmaxf(mxv, sS[row * sp + j]);
    float sum = 0.f;
    for (int j = 0; j < sp; ++j) {
      float ev = (j < s) ? __expf(sS[row * sp + j] - mxv) : 0.f;
      sS[row * sp + j] = ev;
      sum += ev;
    }
    float inv = 1.f / sum;
    for (int j = 0; j < sp; ++j) sQP[row * sp + j] = (_Float16)(sS[row * sp + j] * inv);
  }
  __syncthreads();

  // Phase 3: A = P V (V transposed -> contiguous B), residual add
  for (int jt = 0; jt < njt; ++jt) {
    v8f acc0 = {}, acc1 = {};
    const _Float16* vrow = vb + (long)(jt * 16 + n) * sp;
    for (int p0 = 0; p0 < sp; p0 += 64) {
      __builtin_prefetch(vrow + p0 + 64, 0, 3);
      v16h A0 = frag_contig(&sQP[m * sp + p0], h);
      v16h B0 = frag_contig(vrow + p0, h);
      acc0 = wmma_f16(A0, B0, acc0);
      if (p0 + 32 < sp) {
        v16h A1 = frag_contig(&sQP[m * sp + p0 + 32], h);
        v16h B1 = frag_contig(vrow + p0 + 32, h);
        acc1 = wmma_f16(A1, B1, acc1);
      }
    }
#pragma unroll
    for (int v = 0; v < 8; ++v) {
      int ii = i0 + v + 8 * h, jj = jt * 16 + n;
      if (ii < s && jj < s)
        hout[base + (long)ii * sp + jj] =
            xi[((long)b * s + ii) * s + jj] + acc0[v] + acc1[v];
    }
  }
}

// ---------------------------------------------------------------------------
// Fused expert tail: conv5x5 (1->64, WMMA, K=25 padded to 32) + SiLU kept as
// an 18x18x64 f16 LDS tile, then conv3x3 (64->1).  The 64-channel tensor
// never touches HBM (naively ~2 GB of traffic at s=512 => ~90 us at 23 TB/s,
// >10x the WMMA time, so fusion is mandatory on MI455X).
// ---------------------------------------------------------------------------
__global__ void k_tail(const float* __restrict__ hin, const _Float16* __restrict__ c1wrT,
                       const float* __restrict__ c1b, const float* __restrict__ c2w,
                       const float* __restrict__ c2b, float* __restrict__ yi,
                       int s, int sp, int e) {
  __shared__ float    sin_[22 * 22];
  __shared__ _Float16 smid[18 * 18 * 64];
  int b = blockIdx.x, ty0 = blockIdx.y * 16, tx0 = blockIdx.z * 16;
  int tid = threadIdx.x;

  for (int p = tid; p < 22 * 22; p += 256) {
    int iy = ty0 - 3 + p / 22, ix = tx0 - 3 + p % 22;
    float val = 0.f;
    if ((unsigned)iy < (unsigned)s && (unsigned)ix < (unsigned)s)
      val = hin[(long)b * sp * sp + (long)iy * sp + ix];
    sin_[p] = val;
  }
  __syncthreads();

  const int wave = tid >> 5, L = tid & 31, h = L >> 4, m = L & 15, n = L & 15;
  const _Float16* wB = c1wrT + e * 64 * 32;
  for (int t16 = wave; t16 < 21; t16 += 8) {   // 21 tiles cover 18*18=324 mid pixels
    int pbase = t16 * 16;
    v16h A = build_frag([&](int kk) -> _Float16 {
      if (kk >= 25) return (_Float16)0;
      int p = pbase + m;
      if (p >= 324) p = 0;
      int my = p / 18, mx = p % 18;
      int dy = kk / 5, dx = kk % 5;
      return (_Float16)sin_[(my + dy) * 22 + mx + dx];
    });
#pragma unroll
    for (int t = 0; t < 4; ++t) {
      v16h B = frag_contig(wB + (t * 16 + n) * 32, h);
      v8f acc = {};
      acc = wmma_f16(A, B, acc);
#pragma unroll
      for (int vi = 0; vi < 8; ++vi) {
        int p = pbase + vi + 8 * h;
        if (p < 324) {
          int ch = t * 16 + n;
          float val = acc[vi] + c1b[e * 64 + ch];
          float sg = 1.f / (1.f + __expf(-val));
          smid[p * 64 + ch] = (_Float16)(val * sg);
        }
      }
    }
  }
  __syncthreads();

  int ty = tid >> 4, tx = tid & 15;
  int oy = ty0 + ty, ox = tx0 + tx;
  if (oy < s && ox < s) {
    const float* w = c2w + e * 576;   // (64,3,3)
    float acc = c2b[e];
    for (int ch = 0; ch < 64; ++ch) {
#pragma unroll
      for (int dy = 0; dy < 3; ++dy)
#pragma unroll
        for (int dx = 0; dx < 3; ++dx)
          acc += (float)smid[((ty + dy) * 18 + tx + dx) * 64 + ch] * w[ch * 9 + dy * 3 + dx];
    }
    yi[(long)b * s * s + (long)oy * s + ox] = acc;
  }
}

// ---------------------------------------------------------------------------
// Vertical cubic resize of tmp [B, in_len, 128] to 128 rows, fused with
// out += val * gate_weight[:, e]
// ---------------------------------------------------------------------------
__global__ void k_resize_v_accum(const float* __restrict__ tmp, const float* __restrict__ gw,
                                 float* __restrict__ out, int e, int in_len) {
  int idx = blockIdx.x * 256 + threadIdx.x;
  if (idx >= BATCH * HH * WW) return;
  int xx = idx & 127, yy = (idx >> 7) & 127, b = idx >> 14;
  float inv = (float)in_len / 128.f;
  float ks  = inv > 1.f ? inv : 1.f;
  float c   = (yy + 0.5f) * inv - 0.5f;
  int lo = (int)ceilf(c - 2.f * ks);
  int hi = (int)floorf(c + 2.f * ks);
  float sum = 0.f, wsum = 0.f;
  for (int i = lo; i <= hi; ++i) {
    if ((unsigned)i >= (unsigned)in_len) continue;
    float w = cubicw((i - c) / ks);
    sum += w * tmp[((long)b * in_len + i) * 128 + xx];
    wsum += w;
  }
  out[idx] += (sum / wsum) * gw[(((long)(b * 7 + e)) << 14) + (yy << 7) + xx];
}

// ---------------------------------------------------------------------------
extern "C" void kernel_launch(void* const* d_in, const int* in_sizes, int n_in,
                              void* d_out, int out_size, void* d_ws, size_t ws_size,
                              hipStream_t stream) {
  const float* x   = (const float*)d_in[0];
  const float* dec = (const float*)d_in[1];
  const float* w1  = (const float*)d_in[2];
  const float* b1  = (const float*)d_in[3];
  const float* w2  = (const float*)d_in[4];
  const float* b2  = (const float*)d_in[5];
  const float* w3  = (const float*)d_in[6];
  const float* b3  = (const float*)d_in[7];
  const float* qw  = (const float*)d_in[8];
  const float* qb  = (const float*)d_in[9];
  const float* kw  = (const float*)d_in[10];
  const float* kb  = (const float*)d_in[11];
  const float* vw  = (const float*)d_in[12];
  const float* vb  = (const float*)d_in[13];
  const float* c1w = (const float*)d_in[14];
  const float* c1b = (const float*)d_in[15];
  const float* c2w = (const float*)d_in[16];
  const float* c2b = (const float*)d_in[17];
  float* out = (float*)d_out;

  char* ws = (char*)d_ws;
  size_t o = 0;
  float*    gw_   = (float*)(ws + o);    o += 3670016;   // [8,7,128,128] f32
  _Float16* h1    = (_Float16*)(ws + o); o += 16777216;  // [8,128,128,64] f16
  _Float16* w2rT  = (_Float16*)(ws + o); o += 73728;     // [64,576] f16
  _Float16* c1wrT = (_Float16*)(ws + o); o += 28672;     // [7,64,32] f16
  float*    xi    = (float*)(ws + o);    o += 8388608;   // [8,512,512] f32
  _Float16* qbuf  = (_Float16*)(ws + o); o += 4194304;   // [8,512,512] f16 (padded)
  _Float16* kbuf  = (_Float16*)(ws + o); o += 4194304;
  _Float16* vTbuf = (_Float16*)(ws + o); o += 4194304;   // transposed V
  float*    hbuf  = (float*)(ws + o);    o += 8388608;   // [8,sp,sp] f32
  float*    yibuf = (float*)(ws + o);    o += 8388608;   // [8,s,s] f32
  float*    tmp   = (float*)(ws + o);    o += 2097152;   // resize intermediate

  const int npix = BATCH * HH * WW;          // 131072
  k_init_out<<<(npix + 255) / 256, 256, 0, stream>>>(x, out);
  k_gate_c1<<<(npix + 255) / 256, 256, 0, stream>>>(x, w1, b1, h1);
  k_prep_w2r<<<(576 * 64 + 255) / 256, 256, 0, stream>>>(w2, w2rT);
  k_prep_c1wr<<<(7 * 32 * 64 + 255) / 256, 256, 0, stream>>>(c1w, c1wrT);
  k_gate_c2<<<BATCH * HH * (WW / 16), 32, 0, stream>>>(h1, w2rT, b2, w3, b3, gw_);

  static const int SIZES_[7] = {32, 42, 64, 128, 256, 384, 512};
  for (int i = 0; i < 7; ++i) {
    int s = SIZES_[i];
    int sp = (s + 31) & ~31;   // pad to 32 so every K window stays in-row
    long tot;

    // dec [8,128,128] -> tmp [8,128,s] (horizontal), -> xi [8,s,s] (vertical)
    tot = 8L * 128 * s;
    k_resize1d<<<(unsigned)((tot + 255) / 256), 256, 0, stream>>>(
        dec, tmp, 8, 128, 128, s, 128L * 128, 128L, 1L, 128L * s, (long)s, 1L);
    tot = 8L * s * s;
    k_resize1d<<<(unsigned)((tot + 255) / 256), 256, 0, stream>>>(
        tmp, xi, 8, s, 128, s, 128L * s, 1L, (long)s, (long)s * s, 1L, (long)s);

    tot = 8L * sp * sp;
    k_qkv<<<(unsigned)((tot + 255) / 256), 256, 0, stream>>>(
        xi, qbuf, kbuf, vTbuf, qw, qb, kw, kb, vw, vb, i, s, sp);

    dim3 ga(BATCH, sp / 16);
    k_attn<<<ga, 32, 0, stream>>>(qbuf, kbuf, vTbuf, xi, hbuf, s, sp);

    dim3 gt(BATCH, (s + 15) / 16, (s + 15) / 16);
    k_tail<<<gt, 256, 0, stream>>>(hbuf, c1wrT, c1b, c2w, c2b, yibuf, s, sp, i);

    // yi [8,s,s] -> tmp [8,s,128] (horizontal), then vertical + gated accumulate
    tot = 8L * s * 128;
    k_resize1d<<<(unsigned)((tot + 255) / 256), 256, 0, stream>>>(
        yibuf, tmp, 8, s, s, 128, (long)s * s, (long)s, 1L, (long)s * 128, 128L, 1L);
    k_resize_v_accum<<<(npix + 255) / 256, 256, 0, stream>>>(tmp, gw_, out, i, s);
  }

  (void)in_sizes; (void)n_in; (void)out_size; (void)ws_size;
}
